// MAF_14559939134015
// MI455X (gfx1250) — matrix operand order repaired and checked by reading, not compiled, not run on hardware
//
#include <hip/hip_runtime.h>

#define D_ 128
#define H_ 1024
#define L_ 5
#define B_ 32768
#define CHUNK 16384

typedef _Float16 v8h  __attribute__((ext_vector_type(8)));
typedef _Float16 v16h __attribute__((ext_vector_type(16)));
typedef float    v8f  __attribute__((ext_vector_type(8)));

// Build a 16-half fragment from two contiguous-8 chunks at p and p+gap (16B aligned each).
static __device__ __forceinline__ v16h load_frag2(const _Float16* p, int gap) {
    v8h lo = *(const v8h*)(p);
    v8h hi = *(const v8h*)(p + gap);
    return __builtin_shufflevector(lo, hi, 0,1,2,3,4,5,6,7,8,9,10,11,12,13,14,15);
}

static __device__ __forceinline__ v8f wmma_f16(v16h a, v16h b, v8f c) {
    return __builtin_amdgcn_wmma_f32_16x16x32_f16(false, a, false, b, (short)0, c, false, false);
}

// ---------------- prep: masked, transposed (N-major) f16 weights ----------------
__global__ void k_prep_w1(const float* __restrict__ W1, _Float16* __restrict__ w1t) {
    int idx = blockIdx.x * blockDim.x + threadIdx.x;     // idx = i*H*D + j*D + d
    if (idx >= L_ * H_ * D_) return;
    int i = idx / (H_ * D_);
    int rem = idx - i * (H_ * D_);
    int j = rem / D_;
    int d = rem - j * D_;
    int ord = (i & 1) ? d : (D_ - 1 - d);
    float m = ((j % D_) >= ord) ? 1.f : 0.f;
    float v = W1[(size_t)i * D_ * H_ + (size_t)d * H_ + j] * m;
    w1t[idx] = (_Float16)v;                              // layout [i][j][d]
}

__global__ void k_prep_w2(const float* __restrict__ W2, _Float16* __restrict__ w2t) {
    int idx = blockIdx.x * blockDim.x + threadIdx.x;     // idx = i*256*H + k*H + j
    if (idx >= L_ * 2 * D_ * H_) return;
    int i = idx / (2 * D_ * H_);
    int rem = idx - i * (2 * D_ * H_);
    int k = rem / H_;
    int j = rem - k * H_;
    int kk = k & (D_ - 1);
    int ord = (i & 1) ? kk : (D_ - 1 - kk);
    float m = ((j % D_) >= ord) ? 1.f : 0.f;
    float v = W2[(size_t)i * H_ * (2 * D_) + (size_t)j * (2 * D_) + k] * m;
    w2t[idx] = (_Float16)v;                              // layout [i][k][j]
}

__global__ void k_zero_ldj(float* __restrict__ ldj) {
    int idx = blockIdx.x * blockDim.x + threadIdx.x;
    if (idx < B_) ldj[idx] = 0.f;
}

__global__ void k_conv_x(const float* __restrict__ x, _Float16* __restrict__ xf, int n) {
    int idx = blockIdx.x * blockDim.x + threadIdx.x;
    if (idx < n) xf[idx] = (_Float16)x[idx];
}

// ---------------- GEMM1: h = relu(x @ W1m + b1), LDS-staged ----------------
#define XP 136   // padded halfs per x/W1 LDS row (272B: conflict-free fragment reads)

__global__ __launch_bounds__(256) void k_gemm1(
    const _Float16* __restrict__ xf16,   // CHUNK x 128
    const _Float16* __restrict__ w1t,    // H x D  (N-major masked weights, layer slice)
    const float*    __restrict__ b1,     // H (layer slice)
    _Float16*       __restrict__ h)      // CHUNK x 1024
{
    __shared__ __align__(32) _Float16 sx[64 * XP];    // 17.4 KB
    __shared__ __align__(32) _Float16 sw[128 * XP];   // 34.8 KB

    const int tid = threadIdx.x;
    const int wave = tid >> 5, lane = tid & 31;
    const int mrow = wave & 3, nhalf = wave >> 2;
    const int bm = blockIdx.x, bn = blockIdx.y;
    const int rowA  = lane & 15;
    const int koffA = (lane >> 4) << 3;
    const int nB    = lane & 15;
    const int koffB = (lane >> 4) << 3;

    // stage x tile (64 rows x 128 halfs) and W1 tile (128 rows x 128 halfs)
#pragma unroll
    for (int u = tid; u < 64 * 16; u += 256) {
        int row = u >> 4, c = u & 15;
        *(v8h*)&sx[row * XP + c * 8] =
            *(const v8h*)(xf16 + (size_t)(bm * 64 + row) * D_ + c * 8);
    }
#pragma unroll
    for (int u = tid; u < 128 * 16; u += 256) {
        int row = u >> 4, c = u & 15;
        *(v8h*)&sw[row * XP + c * 8] =
            *(const v8h*)(w1t + (size_t)(bn * 128 + row) * D_ + c * 8);
    }
    __syncthreads();

    const _Float16* xr = sx + (mrow * 16 + rowA) * XP + koffA;
    v16h a[4];
#pragma unroll
    for (int k = 0; k < 4; ++k) a[k] = load_frag2(xr + k * 32, 16);

#pragma unroll
    for (int nt = 0; nt < 4; ++nt) {
        const int lcol = (nhalf * 4 + nt) * 16 + nB;         // 0..127
        const _Float16* wb = sw + lcol * XP + koffB;
        v8f acc = {0.f, 0.f, 0.f, 0.f, 0.f, 0.f, 0.f, 0.f};
#pragma unroll
        for (int k = 0; k < 4; ++k) {
            v16h b = load_frag2(wb + k * 32, 16);
            acc = wmma_f16(a[k], b, acc);
        }
        const int coln = bn * 128 + (nhalf * 4 + nt) * 16 + (lane & 15);
        const float bias = b1[coln];
        const int mb = bm * 64 + mrow * 16 + ((lane >> 4) << 3);
        _Float16* hp = h + (size_t)mb * H_ + coln;
#pragma unroll
        for (int r = 0; r < 8; ++r) {
            float v = acc[r] + bias;
            v = v > 0.f ? v : 0.f;
            hp[(size_t)r * H_] = (_Float16)v;
        }
    }
}

// ------- GEMM2 + epilogue, double-buffered LDS K-slices -------
#define SP 40    // padded halfs per staged K-slice row (80B: conflict-free)

__global__ __launch_bounds__(256) void k_gemm2(
    const _Float16* __restrict__ h,      // CHUNK x 1024
    const _Float16* __restrict__ w2t,    // 256 x 1024 (N-major masked weights, layer slice)
    const float*    __restrict__ b2,     // 256 (layer slice)
    const float*    __restrict__ xin,    // chunk-local rows x 128 f32
    float*          __restrict__ xo32,   // chunk-local rows x 128 f32 (may be d_out region)
    _Float16*       __restrict__ xo16,   // chunk-local rows x 128 f16
    float*          __restrict__ ldj,    // chunk-local rows
    int evenLayer)
{
    __shared__ __align__(32) _Float16 sh2[2][64 * SP];    // 10 KB
    __shared__ __align__(32) _Float16 sw2[2][256 * SP];   // 40 KB
    __shared__ float ldj_lds[64];

    const int tid = threadIdx.x;
    if (tid < 64) ldj_lds[tid] = 0.f;

    const int wave = tid >> 5, lane = tid & 31;
    const int mrow = wave & 3, g = wave >> 2;     // g picks d-range [g*64, g*64+64)
    const int bm = blockIdx.x;
    const int rowA  = lane & 15;
    const int koffA = (lane >> 4) << 3;
    const int nB    = lane & 15;
    const int koffB = (lane >> 4) << 3;

    // staging assignments: one W2 row (tid) = 32 halfs; h rows spread 4 threads/row
    const _Float16* wsrc = w2t + (size_t)tid * H_;
    const _Float16* hsrc = h + (size_t)(bm * 64 + (tid >> 2)) * H_ + (tid & 3) * 8;

    v8h wreg0, wreg1, wreg2, wreg3, hreg;
    auto gload = [&](int k) {
        const _Float16* wp = wsrc + k * 32;
        wreg0 = *(const v8h*)(wp + 0);
        wreg1 = *(const v8h*)(wp + 8);
        wreg2 = *(const v8h*)(wp + 16);
        wreg3 = *(const v8h*)(wp + 24);
        hreg  = *(const v8h*)(hsrc + k * 32);
    };
    auto sstore = [&](int k) {
        _Float16* wd = &sw2[k & 1][tid * SP];
        *(v8h*)(wd + 0)  = wreg0;
        *(v8h*)(wd + 8)  = wreg1;
        *(v8h*)(wd + 16) = wreg2;
        *(v8h*)(wd + 24) = wreg3;
        *(v8h*)(&sh2[k & 1][(tid >> 2) * SP + (tid & 3) * 8]) = hreg;
    };

    v8f accM[4], accL[4];
    const v8f vz = {0.f, 0.f, 0.f, 0.f, 0.f, 0.f, 0.f, 0.f};
#pragma unroll
    for (int p = 0; p < 4; ++p) { accM[p] = vz; accL[p] = vz; }

    gload(0);
    sstore(0);
    __syncthreads();

#pragma unroll 2
    for (int k = 0; k < 32; ++k) {
        if (k + 1 < 32) gload(k + 1);                 // overlap with WMMAs below

        const _Float16* shb = sh2[k & 1];
        const _Float16* swb = sw2[k & 1];
        v16h a = load_frag2(shb + (mrow * 16 + rowA) * SP + koffA, 16);
#pragma unroll
        for (int p = 0; p < 4; ++p) {
            const int dcol = g * 64 + p * 16 + nB;    // mean column (N)
            v16h bM = load_frag2(swb + dcol * SP + koffB, 16);
            accM[p] = wmma_f16(a, bM, accM[p]);
            v16h bL = load_frag2(swb + (D_ + dcol) * SP + koffB, 16);
            accL[p] = wmma_f16(a, bL, accL[p]);
        }

        if (k + 1 < 32) sstore(k + 1);
        __syncthreads();
    }

    // epilogue: mean tile p and log_scale tile p share identical (lane,r)->(m,d) mapping
    const int mloc = mrow * 16 + ((lane >> 4) << 3);   // local row base within 64-row tile
    float tacc[8] = {0.f, 0.f, 0.f, 0.f, 0.f, 0.f, 0.f, 0.f};
#pragma unroll
    for (int p = 0; p < 4; ++p) {
        const int d = g * 64 + p * 16 + (lane & 15);
        const float bMv = b2[d];
        const float bLv = b2[D_ + d];
        const int c = evenLayer ? (D_ - 1 - d) : d;    // new_x[:,c] = z[:,perm[c]]
#pragma unroll
        for (int r = 0; r < 8; ++r) {
            const size_t grow = (size_t)bm * 64 + mloc + r;
            float mean = accM[p][r] + bMv;
            float ls   = accL[p][r] + bLv;
            float t = tanhf(ls);
            float z = (xin[grow * D_ + d] - mean) * __expf(-t);
            xo32[grow * D_ + c] = z;
            xo16[grow * D_ + c] = (_Float16)z;
            tacc[r] += t;
        }
    }
#pragma unroll
    for (int r = 0; r < 8; ++r) {
        float s = tacc[r];
        s += __shfl_xor(s, 1, 32);
        s += __shfl_xor(s, 2, 32);
        s += __shfl_xor(s, 4, 32);
        s += __shfl_xor(s, 8, 32);                     // sum over 16-lane group
        if ((lane & 15) == 0) atomicAdd(&ldj_lds[mloc + r], s);
    }
    __syncthreads();
    if (tid < 64) ldj[(size_t)bm * 64 + tid] -= ldj_lds[tid];
}

extern "C" void kernel_launch(void* const* d_in, const int* in_sizes, int n_in,
                              void* d_out, int out_size, void* d_ws, size_t ws_size,
                              hipStream_t stream) {
    const float* x  = (const float*)d_in[0];
    const float* W1 = (const float*)d_in[1];
    const float* b1 = (const float*)d_in[2];
    const float* W2 = (const float*)d_in[3];
    const float* b2 = (const float*)d_in[4];
    float* out_x   = (float*)d_out;
    float* out_ldj = out_x + (size_t)B_ * D_;

    char* ws = (char*)d_ws;
    size_t off = 0;
    auto carve = [&](size_t bytes) -> char* {
        char* p = ws + off;
        off = (off + bytes + 255) & ~(size_t)255;
        return p;
    };
    _Float16* w1t   = (_Float16*)carve((size_t)L_ * H_ * D_ * 2);
    _Float16* w2t   = (_Float16*)carve((size_t)L_ * 2 * D_ * H_ * 2);
    _Float16* hbuf  = (_Float16*)carve((size_t)CHUNK * H_ * 2);
    float*    xf32a = (float*)   carve((size_t)CHUNK * D_ * 4);
    float*    xf32b = (float*)   carve((size_t)CHUNK * D_ * 4);
    _Float16* xf16a = (_Float16*)carve((size_t)CHUNK * D_ * 2);
    _Float16* xf16b = (_Float16*)carve((size_t)CHUNK * D_ * 2);

    k_prep_w1<<<(L_ * H_ * D_ + 255) / 256, 256, 0, stream>>>(W1, w1t);
    k_prep_w2<<<(L_ * 2 * D_ * H_ + 255) / 256, 256, 0, stream>>>(W2, w2t);
    k_zero_ldj<<<(B_ + 255) / 256, 256, 0, stream>>>(out_ldj);

    for (int c = 0; c < B_ / CHUNK; ++c) {
        const size_t ro = (size_t)c * CHUNK;
        k_conv_x<<<(CHUNK * D_ + 255) / 256, 256, 0, stream>>>(x + ro * D_, xf16a, CHUNK * D_);

        const float* xin = x + ro * D_;
        _Float16* xf16_cur = xf16a;
        float*    f32dst[2] = {xf32a, xf32b};
        _Float16* f16dst[2] = {xf16b, xf16a};

        for (int l = 0; l < L_; ++l) {
            dim3 g1(CHUNK / 64, H_ / 128);
            k_gemm1<<<g1, 256, 0, stream>>>(xf16_cur,
                                            w1t + (size_t)l * H_ * D_,
                                            b1 + (size_t)l * H_, hbuf);
            float*    xo32 = (l == L_ - 1) ? (out_x + ro * D_) : f32dst[l & 1];
            _Float16* xo16 = f16dst[l & 1];
            k_gemm2<<<CHUNK / 64, 256, 0, stream>>>(hbuf,
                                                    w2t + (size_t)l * 2 * D_ * H_,
                                                    b2 + (size_t)l * 2 * D_,
                                                    xin, xo32, xo16, out_ldj + ro,
                                                    (l & 1) == 0 ? 1 : 0);
            xin = xo32;
            xf16_cur = xo16;
        }
    }
    (void)in_sizes; (void)n_in; (void)out_size; (void)ws_size;
}
